// MAGAC_6811818131891
// MI455X (gfx1250) — compile-verified, matrix operand order checked by reading, not run on hardware
//
#include <hip/hip_runtime.h>
#include <hip/hip_bf16.h>
#include <stdint.h>

// ---------------------------------------------------------------------------
// Problem constants (match reference)
// ---------------------------------------------------------------------------
#define NN   4096   // nodes
#define DE   16     // embedding dim
#define NH   4      // heads
#define NK   3      // Chebyshev order (K)
#define LL   64     // in_dim
#define BB   16     // batch
#define NC   (BB * LL)   // 1024 GEMM output columns (b*L + l)

typedef __bf16 v8bf  __attribute__((ext_vector_type(8)));
typedef __bf16 v16bf __attribute__((ext_vector_type(16)));
typedef float  v8f   __attribute__((ext_vector_type(8)));

// ---------------------------------------------------------------------------
// CDNA5 async global->LDS copy (GLOBAL_LOAD_ASYNC_TO_LDS_B128, ASYNCcnt)
// LDS address = low 32 bits of generic pointer (ISA: LDS aperture uses
// addr[31:0]); global address in a 64-bit VGPR pair.
// ---------------------------------------------------------------------------
__device__ __forceinline__ void async_copy16(const __bf16* lds_p, const __bf16* g_p) {
    uint32_t l = (uint32_t)(uintptr_t)lds_p;
    uint64_t g = (uint64_t)(uintptr_t)g_p;
    asm volatile("global_load_async_to_lds_b128 %0, %1, off"
                 :: "v"(l), "v"(g) : "memory");
}

__device__ __forceinline__ void wait_async0() {
#if __has_builtin(__builtin_amdgcn_s_wait_asynccnt)
    __builtin_amdgcn_s_wait_asynccnt(0);
#else
    asm volatile("s_wait_asynccnt 0x0" ::: "memory");
#endif
}

// ---------------------------------------------------------------------------
// Wave / block reduction helpers (wave32, 256-thread blocks = 8 waves)
// ---------------------------------------------------------------------------
__device__ inline float waveMax(float v) {
    #pragma unroll
    for (int o = 16; o > 0; o >>= 1) v = fmaxf(v, __shfl_xor(v, o, 32));
    return v;
}
__device__ inline float waveSum(float v) {
    #pragma unroll
    for (int o = 16; o > 0; o >>= 1) v += __shfl_xor(v, o, 32);
    return v;
}
__device__ inline float blockMax(float v, float* red) {
    v = waveMax(v);
    int wave = threadIdx.x >> 5, lane = threadIdx.x & 31;
    if (lane == 0) red[wave] = v;
    __syncthreads();
    float r = red[0];
    #pragma unroll
    for (int i = 1; i < 8; ++i) r = fmaxf(r, red[i]);
    __syncthreads();
    return r;
}
__device__ inline float blockSum(float v, float* red) {
    v = waveSum(v);
    int wave = threadIdx.x >> 5, lane = threadIdx.x & 31;
    if (lane == 0) red[wave] = v;
    __syncthreads();
    float r = red[0];
    #pragma unroll
    for (int i = 1; i < 8; ++i) r += red[i];
    __syncthreads();
    return r;
}

// ---------------------------------------------------------------------------
// Gaussian adjacency: A_g[i,:] = softmax_j( exp(-psi * ||psi_i - psi_j||^2) )
// ---------------------------------------------------------------------------
__global__ __launch_bounds__(256)
void k_gauss(const float* __restrict__ pe, const float* __restrict__ psi_s,
             float* __restrict__ Ag) {
    __shared__ float red[8];
    const int i = blockIdx.x, tid = threadIdx.x;
    float pi[DE];
    #pragma unroll
    for (int d = 0; d < DE; ++d) pi[d] = pe[i * DE + d];
    const float psi = psi_s[0];

    float ev[NN / 256];
    float lmax = -3.4e38f;
    int vi = 0;
    for (int j = tid; j < NN; j += 256, ++vi) {
        float d2 = 0.f;
        #pragma unroll
        for (int d = 0; d < DE; ++d) {
            float t = pi[d] - pe[j * DE + d];
            d2 = fmaf(t, t, d2);
        }
        float e = __expf(-psi * d2);
        ev[vi] = e;
        lmax = fmaxf(lmax, e);
    }
    float mx = blockMax(lmax, red);
    float ls = 0.f;
    #pragma unroll
    for (int v = 0; v < NN / 256; ++v) ls += __expf(ev[v] - mx);
    float s = blockSum(ls, red);
    float inv = 1.f / s;
    vi = 0;
    for (int j = tid; j < NN; j += 256, ++vi)
        Ag[(size_t)i * NN + j] = __expf(ev[vi] - mx) * inv;
}

// ---------------------------------------------------------------------------
// Q/K projection for all heads: Q[n,h,m] = sum_d pe[n,d] * W_q[d,h,m]
// ---------------------------------------------------------------------------
__global__ __launch_bounds__(256)
void k_qkproj(const float* __restrict__ pe, const float* __restrict__ Wq,
              const float* __restrict__ Wk, float* __restrict__ Qh,
              float* __restrict__ Kh) {
    int idx = blockIdx.x * 256 + threadIdx.x;     // n * (H*DE) + h*DE + m
    if (idx >= NN * NH * DE) return;
    int n  = idx / (NH * DE);
    int hm = idx - n * (NH * DE);                 // h*DE + m
    float q = 0.f, k = 0.f;
    #pragma unroll
    for (int d = 0; d < DE; ++d) {
        float p = pe[n * DE + d];
        q = fmaf(p, Wq[d * (NH * DE) + hm], q);
        k = fmaf(p, Wk[d * (NH * DE) + hm], k);
    }
    Qh[idx] = q;
    Kh[idx] = k;
}

// ---------------------------------------------------------------------------
// Per-head: A_eff(bf16) = alpha * A_g + (1-alpha) * softmax_j(Q_i . K_j / 4)
// ---------------------------------------------------------------------------
__global__ __launch_bounds__(256)
void k_attn_blend(const float* __restrict__ Qh, const float* __restrict__ Kh,
                  const float* __restrict__ Ag, const float* __restrict__ alpha_s,
                  __bf16* __restrict__ Aeff, int h) {
    __shared__ float red[8];
    const int i = blockIdx.x, tid = threadIdx.x;
    float qi[DE];
    #pragma unroll
    for (int d = 0; d < DE; ++d) qi[d] = Qh[i * (NH * DE) + h * DE + d];

    float ev[NN / 256];
    float lmax = -3.4e38f;
    int vi = 0;
    for (int j = tid; j < NN; j += 256, ++vi) {
        float s = 0.f;
        #pragma unroll
        for (int d = 0; d < DE; ++d)
            s = fmaf(qi[d], Kh[j * (NH * DE) + h * DE + d], s);
        s *= 0.25f;   // 1/sqrt(DE)
        ev[vi] = s;
        lmax = fmaxf(lmax, s);
    }
    float mx = blockMax(lmax, red);
    float ls = 0.f;
    #pragma unroll
    for (int v = 0; v < NN / 256; ++v) ls += __expf(ev[v] - mx);
    float s = blockSum(ls, red);
    float inv = 1.f / s;
    float alpha = 1.f / (1.f + __expf(-alpha_s[0]));
    vi = 0;
    for (int j = tid; j < NN; j += 256, ++vi) {
        float p = __expf(ev[vi] - mx) * inv;
        float a = alpha * Ag[(size_t)i * NN + j] + (1.f - alpha) * p;
        Aeff[(size_t)i * NN + j] = (__bf16)a;
    }
}

// ---------------------------------------------------------------------------
// Pack x (B,N,L) f32 -> Y0^T (NC rows, NN cols) bf16,  Y0t[c, n] = x[b, n, l]
// ---------------------------------------------------------------------------
__global__ __launch_bounds__(256)
void k_packx(const float* __restrict__ x, __bf16* __restrict__ Y0t) {
    int idx = blockIdx.x * 256 + threadIdx.x;     // c * NN + n
    if (idx >= NC * NN) return;
    int n = idx & (NN - 1);
    int c = idx >> 12;            // / NN
    int b = c >> 6;               // / LL
    int l = c & (LL - 1);
    Y0t[idx] = (__bf16)x[((size_t)b * NN + n) * LL + l];
}

// ---------------------------------------------------------------------------
// WMMA GEMM + Chebyshev combine:  Dt = scale * (A @ Ycur) + beta * Yprev
// A    : bf16 row-major (NN x NN)
// Bt   : bf16 (NC x NN), B[k][c] == Bt[c*NN + k]  -> contiguous K per lane
// Pt/Dt: bf16 (NC x NN)
// Block tile 256(M) x 128(cols), 256 threads = 8 waves.
// Wave w owns two 16-row strips (w*16 and 128 + w*16): each B fragment feeds
// two WMMAs -> 16 v_wmma per wave per K-step vs 20 ds_load_b128.
// Global->LDS staging uses CDNA5 async loads (no VGPR round trip).
// ---------------------------------------------------------------------------
#define TM  256
#define TNC 128
#define TK  32

__device__ __forceinline__ v16bf ldfrag(const __bf16* p) {  // lo at p, hi at p+16
    v8bf lo = *(const v8bf*)(p);
    v8bf hi = *(const v8bf*)(p + 16);
    v16bf r;
    #pragma unroll
    for (int i = 0; i < 8; ++i) { r[i] = lo[i]; r[i + 8] = hi[i]; }
    return r;
}

__global__ __launch_bounds__(256)
void k_gemm_cheb(const __bf16* __restrict__ A, const __bf16* __restrict__ Bt,
                 const __bf16* __restrict__ Pt, __bf16* __restrict__ Dt,
                 float scale, float beta) {
    __shared__ __align__(16) __bf16 As[TM * TK];    // 16 KB
    __shared__ __align__(16) __bf16 Bs[TNC * TK];   //  8 KB

    const int tid  = threadIdx.x;
    const int wave = tid >> 5;
    const int lane = tid & 31;
    const int l16  = lane & 15;
    const int half = lane >> 4;
    const int kh   = half * 8;          // K-chunk base (lanes 0-15: 0, 16-31: 8)
    const int mBase = blockIdx.x * TM;
    const int cBase = blockIdx.y * TNC;

    v8f acc0[8], acc1[8];
    #pragma unroll
    for (int j = 0; j < 8; ++j)
        #pragma unroll
        for (int r = 0; r < 8; ++r) { acc0[j][r] = 0.f; acc1[j][r] = 0.f; }

    // per-thread staging assignment (constant across K loop):
    // A tile: thread t copies the full 32-element row t (4 x 16B chunks)
    const __bf16* gA = &A[(size_t)(mBase + tid) * NN];
    const __bf16* lA = &As[tid * TK];
    // B tile: thread t copies 2 x 16B chunks of row (t/2)
    const int bc0 = tid * 2;
    const int bR  = bc0 >> 2;
    const int bOff = (bc0 & 3) << 3;
    const __bf16* gB = &Bt[(size_t)(cBase + bR) * NN + bOff];
    const __bf16* lB = &Bs[bR * TK + bOff];

    for (int k0 = 0; k0 < NN; k0 += TK) {
        // --- async global -> LDS staging (ASYNCcnt) ---
        #pragma unroll
        for (int c = 0; c < 4; ++c)
            async_copy16(lA + c * 8, gA + k0 + c * 8);
        async_copy16(lB,     gB + k0);
        async_copy16(lB + 8, gB + k0 + 8);
        wait_async0();
        __syncthreads();

        // --- A fragments for this wave's two 16-row strips ---
        v16bf a0 = ldfrag(&As[(wave * 16 + l16) * TK + kh]);
        v16bf a1 = ldfrag(&As[(128 + wave * 16 + l16) * TK + kh]);

        // --- 8 column sub-tiles, B-frag pipelined one ahead ---
        v16bf bc = ldfrag(&Bs[l16 * TK + kh]);
        #pragma unroll
        for (int j = 0; j < 8; ++j) {
            v16bf bn;
            if (j < 7) bn = ldfrag(&Bs[((j + 1) * 16 + l16) * TK + kh]);
            acc0[j] = __builtin_amdgcn_wmma_f32_16x16x32_bf16(
                false, a0, false, bc, (short)0, acc0[j], false, false);
            acc1[j] = __builtin_amdgcn_wmma_f32_16x16x32_bf16(
                false, a1, false, bc, (short)0, acc1[j], false, false);
            if (j < 7) bc = bn;
        }
        __syncthreads();
    }

    // --- epilogue: D element (m, col); VGPR r -> M = r + 8*half ---
    #pragma unroll
    for (int ms = 0; ms < 2; ++ms) {
        const int m0 = mBase + ms * 128 + wave * 16 + half * 8;
        #pragma unroll
        for (int j = 0; j < 8; ++j) {
            const int col = cBase + j * 16 + l16;
            const size_t base = (size_t)col * NN + m0;  // 8 consecutive M, 16B aligned
            v8bf prev = *(const v8bf*)(&Pt[base]);
            v8f  a    = ms ? acc1[j] : acc0[j];
            v8bf o;
            #pragma unroll
            for (int r = 0; r < 8; ++r)
                o[r] = (__bf16)(scale * a[r] + beta * (float)prev[r]);
            *(v8bf*)(&Dt[base]) = o;
        }
    }
}

// ---------------------------------------------------------------------------
// Per-head filter weights: Wf[n, k, l] = sum_d pe[n,d] * F_w[h, d, k, l]
// ---------------------------------------------------------------------------
__global__ __launch_bounds__(256)
void k_wfilter(const float* __restrict__ pe, const float* __restrict__ Fw,
               float* __restrict__ Wf, int h) {
    int idx = blockIdx.x * 256 + threadIdx.x;     // n*256 + (k*64 + l)
    if (idx >= NN * (NK + 1) * LL) return;
    int n = idx >> 8;
    int r = idx & 255;                            // k*LL + l
    const float* F = Fw + (size_t)h * DE * (NK + 1) * LL;
    float a = 0.f;
    #pragma unroll
    for (int d = 0; d < DE; ++d)
        a = fmaf(pe[n * DE + d], F[d * (NK + 1) * LL + r], a);
    Wf[idx] = a;
}

__global__ __launch_bounds__(256)
void k_bfilt(const float* __restrict__ pe, const float* __restrict__ fb,
             float* __restrict__ bf, int h) {
    int n = blockIdx.x * 256 + threadIdx.x;
    if (n >= NN) return;
    float a = 0.f;
    #pragma unroll
    for (int d = 0; d < DE; ++d)
        a = fmaf(pe[n * DE + d], fb[h * DE + d], a);
    bf[n] = a;
}

// ---------------------------------------------------------------------------
// out[b,n] += mix_w[h] * ( sum_k sum_l Yk^T[b*L+l, n] * Wf[n,k,l] + bfilt[n] )
// One thread per (b,n) -> deterministic accumulation, no atomics.
// ---------------------------------------------------------------------------
__global__ __launch_bounds__(256)
void k_facc(const __bf16* __restrict__ Y0, const __bf16* __restrict__ Y1,
            const __bf16* __restrict__ Y2, const __bf16* __restrict__ Y3,
            const float* __restrict__ Wf, const float* __restrict__ bf,
            const float* __restrict__ head_mix, float* __restrict__ out, int h) {
    int idx = blockIdx.x * 256 + threadIdx.x;     // b * NN + n
    if (idx >= BB * NN) return;
    int n = idx & (NN - 1);
    int b = idx >> 12;

    // softmax over 4-element head_mix (tiny, per-thread)
    float hm[NH];
    float mx = -3.4e38f;
    #pragma unroll
    for (int i = 0; i < NH; ++i) { hm[i] = head_mix[i]; mx = fmaxf(mx, hm[i]); }
    float den = 0.f;
    #pragma unroll
    for (int i = 0; i < NH; ++i) den += __expf(hm[i] - mx);
    float mw = __expf(hm[h] - mx) / den;

    const __bf16* Ys[NK + 1] = {Y0, Y1, Y2, Y3};
    float acc = 0.f;
    #pragma unroll
    for (int k = 0; k <= NK; ++k) {
        const __bf16* Y = Ys[k];
        const float*  W = Wf + (size_t)n * ((NK + 1) * LL) + k * LL;
        for (int l = 0; l < LL; ++l) {
            float y = (float)Y[(size_t)((b << 6) + l) * NN + n];
            acc = fmaf(y, W[l], acc);
        }
    }
    out[idx] += mw * (acc + bf[n]);
}

__global__ __launch_bounds__(256)
void k_zero(float* __restrict__ out, int n) {
    int idx = blockIdx.x * 256 + threadIdx.x;
    if (idx < n) out[idx] = 0.f;
}

// ---------------------------------------------------------------------------
// Launch
// ---------------------------------------------------------------------------
extern "C" void kernel_launch(void* const* d_in, const int* in_sizes, int n_in,
                              void* d_out, int out_size, void* d_ws, size_t ws_size,
                              hipStream_t stream) {
    const float* x        = (const float*)d_in[0];   // (B,N,L)
    const float* psi_emb  = (const float*)d_in[1];   // (N,DE)
    const float* psi      = (const float*)d_in[2];   // scalar
    const float* W_q      = (const float*)d_in[3];   // (DE,H,DE)
    const float* W_k      = (const float*)d_in[4];   // (DE,H,DE)
    const float* a_alpha  = (const float*)d_in[5];   // scalar
    const float* F_w      = (const float*)d_in[6];   // (H,DE,K+1,L)
    const float* f_b      = (const float*)d_in[7];   // (H,DE)
    const float* head_mix = (const float*)d_in[8];   // (H,)
    float* out = (float*)d_out;                      // (B,N)

    // workspace carve-out (256B aligned slabs)
    uint8_t* p = (uint8_t*)d_ws;
    auto take = [&](size_t bytes) {
        uint8_t* r = p;
        p += (bytes + 255) & ~(size_t)255;
        return r;
    };
    float*  Ag    = (float*) take((size_t)NN * NN * 4);        // 67 MB
    __bf16* Aeff  = (__bf16*)take((size_t)NN * NN * 2);        // 33.5 MB
    float*  Qh    = (float*) take((size_t)NN * NH * DE * 4);   // 1 MB
    float*  Kh    = (float*) take((size_t)NN * NH * DE * 4);   // 1 MB
    __bf16* Yt[NK + 1];
    for (int k = 0; k <= NK; ++k)
        Yt[k] = (__bf16*)take((size_t)NC * NN * 2);            // 8.4 MB each
    float*  Wf    = (float*) take((size_t)NN * (NK + 1) * LL * 4); // 4.2 MB
    float*  bfilt = (float*) take((size_t)NN * 4);
    (void)ws_size; (void)in_sizes; (void)n_in; (void)out_size;

    // head-independent prep
    k_zero <<<(BB * NN + 255) / 256, 256, 0, stream>>>(out, BB * NN);
    k_gauss<<<NN, 256, 0, stream>>>(psi_emb, psi, Ag);
    k_qkproj<<<(NN * NH * DE + 255) / 256, 256, 0, stream>>>(psi_emb, W_q, W_k, Qh, Kh);
    k_packx<<<(NC * NN + 255) / 256, 256, 0, stream>>>(x, Yt[0]);

    const dim3 gemmGrid(NN / TM, NC / TNC, 1);   // (16, 8)
    for (int h = 0; h < NH; ++h) {
        k_attn_blend<<<NN, 256, 0, stream>>>(Qh, Kh, Ag, a_alpha, Aeff, h);

        // y1 = A @ y0 ; y2 = 2 A y1 - y0 ; y3 = 2 A y2 - y1
        k_gemm_cheb<<<gemmGrid, 256, 0, stream>>>(Aeff, Yt[0], Yt[0], Yt[1], 1.f,  0.f);
        k_gemm_cheb<<<gemmGrid, 256, 0, stream>>>(Aeff, Yt[1], Yt[0], Yt[2], 2.f, -1.f);
        k_gemm_cheb<<<gemmGrid, 256, 0, stream>>>(Aeff, Yt[2], Yt[1], Yt[3], 2.f, -1.f);

        k_wfilter<<<(NN * (NK + 1) * LL + 255) / 256, 256, 0, stream>>>(psi_emb, F_w, Wf, h);
        k_bfilt  <<<(NN + 255) / 256, 256, 0, stream>>>(psi_emb, f_b, bfilt, h);
        k_facc   <<<(BB * NN + 255) / 256, 256, 0, stream>>>(Yt[0], Yt[1], Yt[2], Yt[3],
                                                             Wf, bfilt, head_mix, out, h);
    }
}
